// GNNModel_41515153883607
// MI455X (gfx1250) — matrix operand order repaired and checked by reading, not compiled, not run on hardware
//
#include <hip/hip_runtime.h>
#include <hip/hip_bf16.h>

typedef __attribute__((ext_vector_type(16))) __bf16 v16bf;
typedef __attribute__((ext_vector_type(8)))  float  v8f;

#define NN 100000
#define EE 1600000
#define GG 512
#define DH 128

// ---------------- degree / normalization ----------------
__global__ void deg_init(float* deg, int n) {
    int t = blockIdx.x * blockDim.x + threadIdx.x;
    if (t < n) deg[t] = 1.0f;                       // self-loop
}
__global__ void deg_edge(const long long* __restrict__ dst, float* __restrict__ deg, int ne) {
    int t = blockIdx.x * blockDim.x + threadIdx.x;
    if (t < ne) atomicAdd(&deg[(int)dst[t]], 1.0f);
}
__global__ void deg_finish(float* deg, int n) {     // deg -> dinv in place
    int t = blockIdx.x * blockDim.x + threadIdx.x;
    if (t < n) deg[t] = rsqrtf(deg[t]);
}

// ---------------- GEMM + pre-scale + self-loop init, via bf16 WMMA ----------------
// HS[n,c]  = (X @ W)[n,c] * dinv[n]          (what edges gather)
// AGG[n,c] = HS[n,c] * dinv[n]               (self-loop contribution, aggregation init)
// Block = 256 threads = 8 waves (one per 16-col tile); each wave does TWO 16-row
// tiles per B fragment (block covers a 32-row strip; N/32 = 3125 exact).
__global__ void gemm_wmma_fused(const float* __restrict__ X, const float* __restrict__ W,
                                const float* __restrict__ dinv,
                                float* __restrict__ HS, float* __restrict__ AGG) {
    const int wave = threadIdx.x >> 5;
    const int lane = threadIdx.x & 31;
    const int half = lane >> 4;
    const int ml   = lane & 15;
    const int tileN = wave;                 // 8 column tiles of 16
    const int col   = tileN * 16 + ml;      // B col for this lane
    const int rowBase = blockIdx.x * 32;    // 2 M-tiles of 16 rows

    v8f acc0 = {}, acc1 = {};
    for (int kk = 0; kk < DH; kk += 32) {
        // B 32x16 bf16 layout: lane(0-15)=N, elem j -> K = (j>>3)*16 + half*8 + (j&7)
        v16bf b;
        const float* wc = W + (size_t)(kk + half * 8) * DH + col;
#pragma unroll
        for (int j = 0; j < 16; ++j)
            b[j] = (__bf16)wc[(size_t)(((j >> 3) << 4) + (j & 7)) * DH];

        // A 16x32 bf16 layout: lane(0-15)=M, elem j -> K = (j>>3)*16 + half*8 + (j&7)
        v16bf a0, a1;
        const float* xr0 = X + (size_t)(rowBase + ml) * DH + kk + half * 8;
        const float* xr1 = xr0 + (size_t)16 * DH;
#pragma unroll
        for (int j = 0; j < 16; ++j) {
            int k = ((j >> 3) << 4) + (j & 7);
            a0[j] = (__bf16)xr0[k];
            a1[j] = (__bf16)xr1[k];
        }
        acc0 = __builtin_amdgcn_wmma_f32_16x16x32_bf16(false, a0, false, b, (short)0, acc0, false, false);
        acc1 = __builtin_amdgcn_wmma_f32_16x16x32_bf16(false, a1, false, b, (short)0, acc1, false, false);
    }
    // C/D f32 16x16 layout: acc[r] -> M = half*8 + r, N = lane&15
#pragma unroll
    for (int r = 0; r < 8; ++r) {
        int orow = rowBase + half * 8 + r;
        float di = dinv[orow];
        float hs = acc0[r] * di;
        HS[(size_t)orow * DH + col]  = hs;
        AGG[(size_t)orow * DH + col] = hs * di;
    }
#pragma unroll
    for (int r = 0; r < 8; ++r) {
        int orow = rowBase + 16 + half * 8 + r;
        float di = dinv[orow];
        float hs = acc1[r] * di;
        HS[(size_t)orow * DH + col]  = hs;
        AGG[(size_t)orow * DH + col] = hs * di;
    }
}

// ---------------- edge scatter: agg[dst] += hs[src] * dinv[dst] ----------------
__global__ void edge_agg(const float* __restrict__ hs, const float* __restrict__ dinv,
                         const long long* __restrict__ src, const long long* __restrict__ dst,
                         float* __restrict__ agg, int ne) {
    int e = blockIdx.x * 8 + (threadIdx.x >> 5);   // one wave per edge
    if (e >= ne) return;
    int lane = threadIdx.x & 31;
    int s = (int)src[e], d = (int)dst[e];
    float nd = dinv[d];
    const float4 vv = *(const float4*)(hs + (size_t)s * DH + lane * 4);
    float* ap = agg + (size_t)d * DH + lane * 4;   // L2-resident atomics
    atomicAdd(ap + 0, vv.x * nd);
    atomicAdd(ap + 1, vv.y * nd);
    atomicAdd(ap + 2, vv.z * nd);
    atomicAdd(ap + 3, vv.w * nd);
}

// ---------------- bias + BN(eval) + ReLU + optional residual ----------------
__global__ void post_bn_relu(const float* __restrict__ agg, const float* __restrict__ cb,
                             const float* __restrict__ g, const float* __restrict__ bt,
                             const float* __restrict__ m, const float* __restrict__ v,
                             const float* __restrict__ resid, float* __restrict__ out, int total) {
    int t = blockIdx.x * blockDim.x + threadIdx.x;
    if (t >= total) return;
    int c = t & (DH - 1);
    float val = agg[t] + cb[c];
    val = (val - m[c]) * rsqrtf(v[c] + 1e-5f) * g[c] + bt[c];
    val = fmaxf(val, 0.0f);
    if (resid) val += resid[t];
    out[t] = val;
}

// ---------------- pooling ----------------
__global__ void pool_init(float* sum, float* mx, float* cnt, int g_) {
    int t = blockIdx.x * blockDim.x + threadIdx.x;
    if (t >= g_ * DH) return;
    sum[t] = 0.0f;
    mx[t]  = 0.0f;                      // features are >= 0, so 0 is a safe identity
    if ((t & (DH - 1)) == 0) cnt[t >> 7] = 0.0f;
}
__global__ void pool_acc(const float* __restrict__ x, const long long* __restrict__ batch,
                         float* __restrict__ sum, float* __restrict__ mx,
                         float* __restrict__ cnt, int n) {
    int node = blockIdx.x * 8 + (threadIdx.x >> 5);
    if (node >= n) return;
    int lane = threadIdx.x & 31;
    int g = (int)batch[node];
    if (lane == 0) atomicAdd(&cnt[g], 1.0f);
    const float4 vv = *(const float4*)(x + (size_t)node * DH + lane * 4);
    float* sp = sum + (size_t)g * DH + lane * 4;
    unsigned* mp = (unsigned*)(mx + (size_t)g * DH + lane * 4);
    atomicAdd(sp + 0, vv.x); atomicAdd(sp + 1, vv.y);
    atomicAdd(sp + 2, vv.z); atomicAdd(sp + 3, vv.w);
    // non-negative floats: IEEE bit pattern is monotonic -> uint atomicMax is exact
    atomicMax(mp + 0, __float_as_uint(vv.x));
    atomicMax(mp + 1, __float_as_uint(vv.y));
    atomicMax(mp + 2, __float_as_uint(vv.z));
    atomicMax(mp + 3, __float_as_uint(vv.w));
}
__global__ void pool_cat(const float* __restrict__ sum, const float* __restrict__ mx,
                         const float* __restrict__ cnt, float* __restrict__ hcat, int g_) {
    int t = blockIdx.x * blockDim.x + threadIdx.x;
    if (t >= g_ * 2 * DH) return;
    int g = t >> 8, c = t & 255;
    hcat[t] = (c < DH) ? sum[g * DH + c] / cnt[g] : mx[g * DH + (c - DH)];
}

// ---------------- tiny MLP head ----------------
__global__ void mlp_fc(const float* __restrict__ in, const float* __restrict__ Wm,
                       const float* __restrict__ bm, float* __restrict__ out,
                       int rows, int K, int ncol, int do_relu) {
    int t = blockIdx.x * blockDim.x + threadIdx.x;
    if (t >= rows * ncol) return;
    int r = t / ncol, o = t - r * ncol;
    float acc = bm[o];
    const float* ir = in + (size_t)r * K;
    for (int k = 0; k < K; ++k)
        acc = fmaf(ir[k], Wm[(size_t)k * ncol + o], acc);
    if (do_relu) acc = fmaxf(acc, 0.0f);
    out[t] = acc;
}

extern "C" void kernel_launch(void* const* d_in, const int* in_sizes, int n_in,
                              void* d_out, int out_size, void* d_ws, size_t ws_size,
                              hipStream_t stream) {
    const float*     x     = (const float*)d_in[0];
    const long long* ei    = (const long long*)d_in[1];   // [2,E] int64
    const long long* batch = (const long long*)d_in[2];   // [N]  int64
    const float* W[3]  = {(const float*)d_in[3],  (const float*)d_in[9],  (const float*)d_in[15]};
    const float* cb[3] = {(const float*)d_in[4],  (const float*)d_in[10], (const float*)d_in[16]};
    const float* gm[3] = {(const float*)d_in[5],  (const float*)d_in[11], (const float*)d_in[17]};
    const float* bt[3] = {(const float*)d_in[6],  (const float*)d_in[12], (const float*)d_in[18]};
    const float* mn[3] = {(const float*)d_in[7],  (const float*)d_in[13], (const float*)d_in[19]};
    const float* vr[3] = {(const float*)d_in[8],  (const float*)d_in[14], (const float*)d_in[20]};
    const float* L1w = (const float*)d_in[21]; const float* L1b = (const float*)d_in[22];
    const float* L2w = (const float*)d_in[23]; const float* L2b = (const float*)d_in[24];
    const float* L3w = (const float*)d_in[25]; const float* L3b = (const float*)d_in[26];

    const long long* src = ei;
    const long long* dst = ei + EE;

    // workspace carve-up (floats)
    float* ws = (float*)d_ws;
    size_t off = 0;
    float* dinv = ws + off; off += NN;
    float* buf0 = ws + off; off += (size_t)NN * DH;
    float* buf1 = ws + off; off += (size_t)NN * DH;
    float* buf2 = ws + off; off += (size_t)NN * DH;
    float* psum = ws + off; off += (size_t)GG * DH;
    float* pmax = ws + off; off += (size_t)GG * DH;
    float* pcnt = ws + off; off += GG;
    float* hcat = ws + off; off += (size_t)GG * 2 * DH;
    float* h1   = ws + off; off += (size_t)GG * DH;
    float* h2   = ws + off; off += (size_t)GG * (DH / 2);
    (void)ws_size; (void)in_sizes; (void)n_in; (void)out_size;

    const int B = 256;
    const int totNF = NN * DH;

    // normalization vector (shared by all layers)
    deg_init<<<(NN + B - 1) / B, B, 0, stream>>>(dinv, NN);
    deg_edge<<<(EE + B - 1) / B, B, 0, stream>>>(dst, dinv, EE);
    deg_finish<<<(NN + B - 1) / B, B, 0, stream>>>(dinv, NN);

    // ---- layer 0: x_in -> buf2 ----
    gemm_wmma_fused<<<NN / 32, B, 0, stream>>>(x, W[0], dinv, buf0, buf1);
    edge_agg<<<(EE + 7) / 8, B, 0, stream>>>(buf0, dinv, src, dst, buf1, EE);
    post_bn_relu<<<(totNF + B - 1) / B, B, 0, stream>>>(buf1, cb[0], gm[0], bt[0], mn[0], vr[0],
                                                        nullptr, buf2, totNF);
    // ---- layer 1: buf2 -> buf0 (residual buf2) ----
    gemm_wmma_fused<<<NN / 32, B, 0, stream>>>(buf2, W[1], dinv, buf0, buf1);
    edge_agg<<<(EE + 7) / 8, B, 0, stream>>>(buf0, dinv, src, dst, buf1, EE);
    post_bn_relu<<<(totNF + B - 1) / B, B, 0, stream>>>(buf1, cb[1], gm[1], bt[1], mn[1], vr[1],
                                                        buf2, buf0, totNF);
    // ---- layer 2: buf0 -> buf1 (residual buf0) ----
    gemm_wmma_fused<<<NN / 32, B, 0, stream>>>(buf0, W[2], dinv, buf1, buf2);
    edge_agg<<<(EE + 7) / 8, B, 0, stream>>>(buf1, dinv, src, dst, buf2, EE);
    post_bn_relu<<<(totNF + B - 1) / B, B, 0, stream>>>(buf2, cb[2], gm[2], bt[2], mn[2], vr[2],
                                                        buf0, buf1, totNF);

    // ---- pooling: mean + max per graph ----
    pool_init<<<(GG * DH + B - 1) / B, B, 0, stream>>>(psum, pmax, pcnt, GG);
    pool_acc<<<(NN + 7) / 8, B, 0, stream>>>(buf1, batch, psum, pmax, pcnt, NN);
    pool_cat<<<(GG * 2 * DH + B - 1) / B, B, 0, stream>>>(psum, pmax, pcnt, hcat, GG);

    // ---- MLP head ----
    mlp_fc<<<(GG * DH + B - 1) / B, B, 0, stream>>>(hcat, L1w, L1b, h1, GG, 2 * DH, DH, 1);
    mlp_fc<<<(GG * (DH / 2) + B - 1) / B, B, 0, stream>>>(h1, L2w, L2b, h2, GG, DH, DH / 2, 1);
    mlp_fc<<<(GG + B - 1) / B, B, 0, stream>>>(h2, L3w, L3b, (float*)d_out, GG, DH / 2, 1, 0);
}
